// LearnableSeriesDecomposition_28561532518484
// MI455X (gfx1250) — compile-verified, matrix operand order checked by reading, not run on hardware
//
#include <hip/hip_runtime.h>
#include <hip/hip_bf16.h>
#include <math.h>

#define B_   64
#define L_   4096
#define D_   128
#define KMAX_ 50
#define H_   64
#define P_   24
#define KW_  49          // 2*P+1
#define TL_  128         // output rows per conv tile
#define HALO_ (TL_ + 2*P_)  // 176 staged rows
#define RW_  8           // sliding-window rows per thread per group

typedef __attribute__((ext_vector_type(16))) _Float16 v16h;
typedef __attribute__((ext_vector_type(8)))  float    v8f;
typedef __attribute__((ext_vector_type(2)))  float    v2f;

// 16-byte int vector type matching the async-builtin's parameter type
typedef int v4i_vs __attribute__((vector_size(16)));
typedef __attribute__((address_space(1))) v4i_vs* gv4p;   // global (AS1)
typedef __attribute__((address_space(3))) v4i_vs* lv4p;   // LDS (AS3)

#if defined(__has_builtin)
#if __has_builtin(__builtin_amdgcn_global_load_async_to_lds_b128)
#define HAVE_ASYNC_B128 1
#endif
#if __has_builtin(__builtin_amdgcn_s_wait_asynccnt)
#define HAVE_WAIT_ASYNC 1
#endif
#endif

__device__ __forceinline__ void async_copy16(const float* g, float* l) {
#if defined(HAVE_ASYNC_B128)
  // (global src v4i AS1*, lds dst v4i AS3*, imm offset, cpol)
  __builtin_amdgcn_global_load_async_to_lds_b128((gv4p)(g), (lv4p)(l), 0, 0);
#else
  unsigned lds_off = (unsigned)(size_t)((__attribute__((address_space(3))) float*)l);
  asm volatile("global_load_async_to_lds_b128 %0, %1, off"
               :: "v"(lds_off), "v"(g) : "memory");
#endif
}

__device__ __forceinline__ void wait_async0() {
#if defined(HAVE_WAIT_ASYNC)
  __builtin_amdgcn_s_wait_asynccnt(0);
#else
  asm volatile("s_wait_asynccnt 0" ::: "memory");
#endif
}

// ---------------- K1: partial sums of x over L (per batch, 8 chunks) -------
__global__ __launch_bounds__(128)
void k_partial_mean(const float* __restrict__ x, float* __restrict__ psum) {
  const int chunk = blockIdx.x;        // 0..7
  const int b     = blockIdx.y;        // 0..63
  const int t     = threadIdx.x;       // 0..127
  const int lane  = t & 31;            // float4 column group (4 channels)
  const int sub   = t >> 5;            // 0..3 row phase
  const float4* xp = (const float4*)(x + (size_t)b * L_ * D_);
  float4 s; s.x = s.y = s.z = s.w = 0.f;
  const int l0 = chunk * 512;
  for (int i = 0; i < 128; ++i) {
    int l = l0 + sub + 4 * i;
    float4 v = xp[(size_t)l * 32 + lane];
    s.x += v.x; s.y += v.y; s.z += v.z; s.w += v.w;
  }
  __shared__ float4 sp[128];
  sp[t] = s;
  __syncthreads();
  if (t < 32) {
    float4 a = sp[t], bb = sp[t + 32], c = sp[t + 64], d = sp[t + 96];
    float4 r;
    r.x = a.x + bb.x + c.x + d.x;
    r.y = a.y + bb.y + c.y + d.y;
    r.z = a.z + bb.z + c.z + d.z;
    r.w = a.w + bb.w + c.w + d.w;
    ((float4*)(psum + (size_t)(b * 8 + chunk) * D_))[t] = r;
  }
}

// ---------------- K2: gate MLP via WMMA, produce k/p per batch -------------
__global__ __launch_bounds__(256)
void k_gate(const float* __restrict__ psum,
            const float* __restrict__ w1, const float* __restrict__ b1,
            const float* __restrict__ w2, const float* __restrict__ b2,
            int* __restrict__ kArr, int* __restrict__ pArr) {
  __shared__ _Float16 Ah[B_][D_];   // x_global as f16 (A matrix, 64x128)
  __shared__ _Float16 Bh[D_][H_];   // w1 as f16      (B matrix, 128x64)
  __shared__ float    Hs[B_][H_];   // hidden (relu'd)
  const int t = threadIdx.x;

  for (int i = t; i < B_ * D_; i += 256) {
    int b = i / D_, d = i % D_;
    float s = 0.f;
    #pragma unroll
    for (int c = 0; c < 8; ++c) s += psum[(size_t)(b * 8 + c) * D_ + d];
    Ah[b][d] = (_Float16)(s * (1.0f / (float)L_));
  }
  for (int i = t; i < D_ * H_; i += 256)
    Bh[i / H_][i % H_] = (_Float16)w1[i];
  __syncthreads();

  const int wv = t >> 5, lane = t & 31;
  const int mt = wv & 3;              // M tile 0..3
  const int nt0 = (wv >> 2) * 2;      // N tile base
  const int half = lane >> 4;
  const int mrow = mt * 16 + (lane & 15);
  for (int ntc = 0; ntc < 2; ++ntc) {
    const int nt = nt0 + ntc;
    const int ncol = nt * 16 + (lane & 15);
    v8f acc = {};
    for (int ks = 0; ks < 4; ++ks) {
      v16h af, bf;
      #pragma unroll
      for (int v = 0; v < 8; ++v) {
        // 16-bit A 16x32 layout: VGPR v holds K pair; lanes16-31 are K+8
        int K0 = ks * 32 + (v >> 2) * 16 + (v & 3) * 2 + half * 8;
        af[2 * v]     = Ah[mrow][K0];
        af[2 * v + 1] = Ah[mrow][K0 + 1];
        bf[2 * v]     = Bh[K0][ncol];
        bf[2 * v + 1] = Bh[K0 + 1][ncol];
      }
      acc = __builtin_amdgcn_wmma_f32_16x16x32_f16(
          false, af, false, bf, (short)0, acc, false, false);
    }
    #pragma unroll
    for (int i = 0; i < 8; ++i) {
      int m = mt * 16 + i + 8 * half;
      float hv = acc[i] + b1[ncol];
      Hs[m][ncol] = hv > 0.f ? hv : 0.f;
    }
  }
  __syncthreads();

  if (t < B_) {
    float dot = b2[0];
    #pragma unroll 8
    for (int h = 0; h < H_; ++h) dot += Hs[t][h] * w2[h];
    float logit = 1.f / (1.f + expf(-dot));
    float kf = logit * (float)(KMAX_ - 5) + 5.f;
    int k = (int)rintf(kf);                 // round-to-nearest-even like jnp.round
    k = k < 3 ? 3 : (k > KMAX_ ? KMAX_ : k);
    if ((k & 1) == 0) k -= 1;
    if (k < 3) k = 3;
    kArr[t] = k;
    pArr[t] = k >> 1;
  }
}

// ---------------- K3: masked softmax -> dense centered taps wo[b][m][d] ----
__global__ __launch_bounds__(128)
void k_weights(const float* __restrict__ tw,
               const int* __restrict__ kArr, const int* __restrict__ pArr,
               float* __restrict__ wo) {
  const int b = blockIdx.x;
  const int d = threadIdx.x;
  const int k = kArr[b], p = pArr[b];
  const float* lg = tw + (size_t)d * KMAX_;   // trend_weights (D,1,KMAX)
  float mx = -3.4e38f;
  for (int tt = 0; tt < k; ++tt) mx = fmaxf(mx, lg[tt]);
  float sum = 0.f;
  for (int tt = 0; tt < k; ++tt) sum += expf(lg[tt] - mx);
  const float inv = 1.f / sum;
  for (int m = 0; m < KW_; ++m) {
    int j = m - P_ + p;
    float v = (j >= 0 && j < k) ? expf(lg[j] - mx) * inv : 0.f;
    wo[((size_t)b * KW_ + m) * D_ + d] = v;
  }
}

// ---------------- K4: depthwise conv (trend) + seasonal --------------------
// Thread = channel pair (v2f); 8-row sliding window; NT stores for outputs.
__global__ __launch_bounds__(128)
void k_conv(const float* __restrict__ x, const float* __restrict__ wo,
            float* __restrict__ seasonal, float* __restrict__ trend) {
  __shared__ float xt[HALO_][D_];              // 176*128*4 = 90112 B
  const int tile = blockIdx.x;                 // 0..31
  const int b    = blockIdx.y;                 // 0..63
  const int t    = threadIdx.x;                // 0..127
  const int l0   = tile * TL_;
  const float* xb = x + (size_t)b * L_ * D_;

  // Async-stage 176 rows (edge-clamped) into LDS; 16B per lane per issue.
  for (int it = 0; it < (HALO_ * 32) / 128; ++it) {   // 44 iterations
    int e  = it * 128 + t;
    int lr = e >> 5;
    int q  = (e & 31) * 4;
    int gl = l0 - P_ + lr;
    gl = gl < 0 ? 0 : (gl > L_ - 1 ? L_ - 1 : gl);
    async_copy16(xb + (size_t)gl * D_ + q, &xt[lr][q]);
  }

  const int cp = (t & 63) * 2;     // channel pair base: 0,2,...,126
  const int hf = t >> 6;           // row half: 0 -> rows 0..63, 1 -> rows 64..127

  // 49 taps (x2 channels) in registers, compile-time indexed only.
  v2f w[KW_];
  #pragma unroll
  for (int m = 0; m < KW_; ++m)
    w[m] = *(const v2f*)(wo + ((size_t)b * KW_ + m) * D_ + cp);

  wait_async0();
  __syncthreads();

  for (int g = 0; g < 64 / RW_; ++g) {
    const int r0 = hf * 64 + g * RW_;
    v2f acc[RW_];
    #pragma unroll
    for (int a = 0; a < RW_; ++a) acc[a] = (v2f){0.f, 0.f};
    #pragma unroll
    for (int i = 0; i < KW_ + RW_ - 1; ++i) {        // 56 b64 LDS reads
      v2f xv = *(const v2f*)&xt[r0 + i][cp];
      #pragma unroll
      for (int a = 0; a < RW_; ++a) {
        int m = i - a;
        if (m >= 0 && m < KW_) acc[a] += w[m] * xv;  // 392 v2f FMAs
      }
    }
    #pragma unroll
    for (int a = 0; a < RW_; ++a) {
      const int r = r0 + a;
      const size_t base = ((size_t)b * L_ + (l0 + r)) * D_ + cp;
      v2f xv = *(const v2f*)&xt[r + P_][cp];
      v2f tr = acc[a];
      v2f se = xv - tr;
      __builtin_nontemporal_store(tr, (v2f*)(trend + base));
      __builtin_nontemporal_store(se, (v2f*)(seasonal + base));
    }
  }
}

extern "C" void kernel_launch(void* const* d_in, const int* in_sizes, int n_in,
                              void* d_out, int out_size, void* d_ws, size_t ws_size,
                              hipStream_t stream) {
  const float* x  = (const float*)d_in[0];
  const float* tw = (const float*)d_in[1];
  const float* w1 = (const float*)d_in[2];
  const float* b1 = (const float*)d_in[3];
  const float* w2 = (const float*)d_in[4];
  const float* b2 = (const float*)d_in[5];

  float* ws   = (float*)d_ws;
  float* psum = ws;                              // 64*8*128 = 65536 floats
  int*   kArr = (int*)(ws + 65536);              // 64 ints
  int*   pArr = kArr + 64;                       // 64 ints
  float* wo   = ws + 65536 + 128;                // 64*49*128 = 401408 floats

  float* seasonal = (float*)d_out;
  float* trend    = seasonal + (size_t)B_ * L_ * D_;

  k_partial_mean<<<dim3(8, 64), 128, 0, stream>>>(x, psum);
  k_gate<<<1, 256, 0, stream>>>(psum, w1, b1, w2, b2, kArr, pArr);
  k_weights<<<64, 128, 0, stream>>>(tw, kArr, pArr, wo);
  k_conv<<<dim3(32, 64), 128, 0, stream>>>(x, wo, seasonal, trend);
}
